// DualFFN_661424963641
// MI455X (gfx1250) — matrix operand order repaired and checked by reading, not compiled, not run on hardware
//
#include <hip/hip_runtime.h>
#include <hip/hip_bf16.h>

// ---------------------------------------------------------------------------
// DualFFN MoE (2 experts, capacity routing) for MI455X / gfx1250.
// bf16 WMMA (v_wmma_f32_16x16x32_bf16) with f32 accumulation.
// M=32 token tiles, 32 waves/block, register-blocked so phase1 is
// 2 loads/WMMA and phase2 is 1 global load/WMMA. Fast-rcp silu epilogue.
// ---------------------------------------------------------------------------

#define TOKENS   8192
#define DIMK     2048
#define HSMALL   2048
#define HLARGE   8192
#define CAP      4096          // ceil(8192/2 * 1.0)

typedef __bf16 bf16_t;
typedef __attribute__((ext_vector_type(16))) __bf16 v16bf;
typedef __attribute__((ext_vector_type(8)))  __bf16 v8bf;
typedef __attribute__((ext_vector_type(8)))  float  v8f;

// ---------------- helpers ----------------

__device__ inline unsigned short f2bf_rne(float f) {
    unsigned u = __builtin_bit_cast(unsigned, f);
    u += 0x7fffu + ((u >> 16) & 1u);
    return (unsigned short)(u >> 16);
}

// silu(g) * u with fast hardware reciprocal (v_rcp_f32), no IEEE div expansion
__device__ inline float silu_mul(float g, float u) {
    float s = __builtin_amdgcn_rcpf(1.f + __expf(-g));
    return g * s * u;
}

__device__ inline v16bf combine16(v8bf lo, v8bf hi) {
    v16bf r;
#pragma unroll
    for (int i = 0; i < 8; ++i) { r[i] = lo[i]; r[i + 8] = hi[i]; }
    return r;
}

// A operand (16x32 bf16, MxK): lane<16 holds K {k0..k0+7, k0+16..k0+23},
// lane>=16 holds K {k0+8..k0+15, k0+24..k0+31}. rowptr = &A[row][0].
__device__ inline v16bf load_opA(const bf16_t* rowptr, int k0, int sel) {
    const bf16_t* p = rowptr + k0 + sel * 8;
    v8bf lo = *(const v8bf*)(p);
    v8bf hi = *(const v8bf*)(p + 16);
    return combine16(lo, hi);
}

// B operand (32x16 bf16, KxN), loaded from row-major [N x K] weight rows:
// lane<16 holds K {k0..k0+7, k0+16..k0+23}, lane>=16 holds K {k0+8..k0+15, k0+24..k0+31}.
__device__ inline v16bf load_opB(const bf16_t* rowptr, int k0, int sel) {
    const bf16_t* p = rowptr + k0 + sel * 8;
    v8bf lo = *(const v8bf*)(p);
    v8bf hi = *(const v8bf*)(p + 16);
    return combine16(lo, hi);
}

__device__ inline v8f wmma_bf16(v16bf a, v16bf b, v8f c) {
    // (neg_a, A, neg_b, B, c_mod, C, reuse_a, reuse_b)
    return __builtin_amdgcn_wmma_f32_16x16x32_bf16(false, a, false, b,
                                                   (short)0, c, false, false);
}

// ---------------- kernel 1: f32 -> bf16 conversion ----------------

__global__ void cvt_kernel(const float* __restrict__ s,
                           unsigned short* __restrict__ d, int n) {
    int i = (blockIdx.x * blockDim.x + threadIdx.x) * 4;
    if (i >= n) return;
    float4 v = *(const float4*)(s + i);
    ushort4 o;
    o.x = f2bf_rne(v.x); o.y = f2bf_rne(v.y);
    o.z = f2bf_rne(v.z); o.w = f2bf_rne(v.w);
    *(ushort4*)(d + i) = o;
}

// ---------------- kernel 2: router (one wave32 per token) ----------------

__global__ void router_kernel(const float* __restrict__ x,
                              const float* __restrict__ rw,
                              const float* __restrict__ rb,
                              int* __restrict__ is_large) {
    int wave = threadIdx.x >> 5;
    int lane = threadIdx.x & 31;
    int token = blockIdx.x * 8 + wave;
    const float* xr = x + (size_t)token * DIMK;
    float s0 = 0.f, s1 = 0.f;
    for (int k = lane; k < DIMK; k += 32) {
        float xv = xr[k];
        s0 += xv * rw[k];
        s1 += xv * rw[DIMK + k];
    }
#pragma unroll
    for (int off = 16; off > 0; off >>= 1) {
        s0 += __shfl_xor(s0, off, 32);
        s1 += __shfl_xor(s1, off, 32);
    }
    if (lane == 0) {
        float l0 = s0 + rb[0];
        float l1 = s1 + rb[1];
        // argmax with first-index tie-break => large iff l1 > l0
        is_large[token] = (l1 > l0) ? 1 : 0;
    }
}

// ---------------- kernel 3: capacity scan + compaction (single block) -----

__global__ void scan_kernel(const int* __restrict__ is_large,
                            int* __restrict__ small_idx,
                            int* __restrict__ large_idx,
                            int* __restrict__ counts,
                            int* __restrict__ out_tail) {
    __shared__ int sh[256];
    const int t = threadIdx.x;
    const int base = t * 32;            // 256 threads * 32 tokens = 8192
    int c = 0;
#pragma unroll 4
    for (int j = 0; j < 32; ++j) c += is_large[base + j];
    sh[t] = c;
    __syncthreads();
    // inclusive scan over 256
    for (int off = 1; off < 256; off <<= 1) {
        int mine = sh[t];
        int add  = (t >= off) ? sh[t - off] : 0;
        __syncthreads();
        sh[t] = mine + add;
        __syncthreads();
    }
    const int totalLarge = sh[255];
    const int excl = sh[t] - c;         // larges strictly before my chunk
    int rank = excl;
    int keptBefore = min(excl, CAP);
    for (int j = 0; j < 32; ++j) {
        int tok = base + j;
        int il = is_large[tok];
        int kept = il && (rank < CAP);
        if (kept) {
            large_idx[keptBefore] = tok;    // keptBefore == rank < CAP
            keptBefore++;
        } else {
            small_idx[tok - keptBefore] = tok;
        }
        rank += il;
    }
    if (t == 0) {
        int rl = min(totalLarge, CAP);
        int rs = TOKENS - rl;
        counts[0] = rs;
        counts[1] = rl;
        out_tail[0] = rs;       // routed_small
        out_tail[1] = rl;       // routed_large
        out_tail[2] = 0;        // dropped
    }
}

// ---------------- kernel 4: fused SwiGLU FFN over a compacted token list --
//
// Block = 1024 threads = 32 waves, one 32-token tile per block.
// Loop over H in 512-wide chunks:
//   phase1: wave w computes act[32 x 16] for h-cols [hc+16w, hc+16w+16):
//           g = x@Wg^T, u = x@Wu^T over K=2048 with two 16-row A sub-tiles
//           sharing each weight B-tile (4 WMMA per 8 b128 loads),
//           silu(g)*u -> bf16 into padded LDS tile act[32][520].
//   phase2: wave w owns out cols [64w, 64w+64): 8 f32 WMMA accumulators
//           (4 col-tiles x 2 token sub-tiles), each wd B-tile load feeds
//           two WMMAs (1 global load / WMMA).

template <int H>
__global__ __launch_bounds__(1024)
void ffn_kernel(const bf16_t* __restrict__ xb,     // [TOKENS, DIMK]
                const bf16_t* __restrict__ wg,     // [H, DIMK]
                const bf16_t* __restrict__ wu,     // [H, DIMK]
                const bf16_t* __restrict__ wd,     // [DIMK, H]
                const int*    __restrict__ idx,
                const int*    __restrict__ count_ptr,
                float*        __restrict__ out) {  // [TOKENS, DIMK]
    constexpr int HC   = 512;
    constexpr int LSTR = 520;   // halves; rows stay 16B-aligned, banks rotate
    __shared__ unsigned short act[32 * LSTR];
    __shared__ int toks[32];

    const int count = *count_ptr;
    const int tile  = blockIdx.x;
    if (tile * 32 >= count) return;

    const int wave = threadIdx.x >> 5;
    const int lane = threadIdx.x & 31;
    const int lr   = lane & 15;
    const int sel  = lane >> 4;

    if (threadIdx.x < 32) {
        int gi = tile * 32 + threadIdx.x;
        toks[threadIdx.x] = (gi < count) ? idx[gi] : 0;
    }
    __syncthreads();

    const bf16_t* xrow0 = xb + (size_t)toks[lr]      * DIMK;  // A rows, sub 0
    const bf16_t* xrow1 = xb + (size_t)toks[16 + lr] * DIMK;  // A rows, sub 1

    v8f oacc[8];    // [0..3] = token sub-tile 0, [4..7] = sub-tile 1
#pragma unroll
    for (int t = 0; t < 8; ++t)
        oacc[t] = (v8f){0.f, 0.f, 0.f, 0.f, 0.f, 0.f, 0.f, 0.f};

    for (int hc = 0; hc < H; hc += HC) {
        // ---- phase 1: this wave's 16 hidden columns, both token sub-tiles
        const int hb = hc + wave * 16;
        const bf16_t* gr = wg + (size_t)(hb + lr) * DIMK;
        const bf16_t* ur = wu + (size_t)(hb + lr) * DIMK;
        v8f g0 = (v8f){0.f, 0.f, 0.f, 0.f, 0.f, 0.f, 0.f, 0.f};
        v8f g1 = g0, u0 = g0, u1 = g0;
#pragma unroll 2
        for (int k0 = 0; k0 < DIMK; k0 += 32) {
            v16bf a0 = load_opA(xrow0, k0, sel);
            v16bf a1 = load_opA(xrow1, k0, sel);
            v16bf bg = load_opB(gr,    k0, sel);
            v16bf bu = load_opB(ur,    k0, sel);
            g0 = wmma_bf16(a0, bg, g0);
            g1 = wmma_bf16(a1, bg, g1);
            u0 = wmma_bf16(a0, bu, u0);
            u1 = wmma_bf16(a1, bu, u1);
        }
        // silu(g)*u -> LDS act tile (C layout: M = v + 8*sel, N = lr)
        const int col = wave * 16 + lr;
#pragma unroll
        for (int v = 0; v < 8; ++v) {
            int m = v + sel * 8;
            act[m * LSTR + col]        = f2bf_rne(silu_mul(g0[v], u0[v]));
            act[(16 + m) * LSTR + col] = f2bf_rne(silu_mul(g1[v], u1[v]));
        }
        __syncthreads();

        // ---- phase 2: contract act tile against wd (cols [64w, 64w+64))
        const int cb = wave * 64;
#pragma unroll
        for (int k0 = 0; k0 < HC; k0 += 32) {
            const bf16_t* ar0 = (const bf16_t*)&act[lr * LSTR];
            const bf16_t* ar1 = (const bf16_t*)&act[(16 + lr) * LSTR];
            v16bf a0 = load_opA(ar0, k0, sel);
            v16bf a1 = load_opA(ar1, k0, sel);
#pragma unroll
            for (int t = 0; t < 4; ++t) {
                const bf16_t* dr = wd + (size_t)(cb + t * 16 + lr) * H + hc;
                v16bf b = load_opB(dr, k0, sel);
                oacc[t]     = wmma_bf16(a0, b, oacc[t]);
                oacc[4 + t] = wmma_bf16(a1, b, oacc[4 + t]);
            }
        }
        __syncthreads();
    }

    // ---- writeback (C layout: element (m = v + 8*sel, n = lr))
#pragma unroll
    for (int sub = 0; sub < 2; ++sub) {
#pragma unroll
        for (int t = 0; t < 4; ++t) {
            int cb = wave * 64 + t * 16;
#pragma unroll
            for (int v = 0; v < 8; ++v) {
                int m  = v + sel * 8 + sub * 16;
                int gi = tile * 32 + m;
                if (gi < count) {
                    out[(size_t)toks[m] * DIMK + cb + lr] = oacc[sub * 4 + t][v];
                }
            }
        }
    }
}

// ---------------- launch ----------------

extern "C" void kernel_launch(void* const* d_in, const int* in_sizes, int n_in,
                              void* d_out, int out_size, void* d_ws, size_t ws_size,
                              hipStream_t stream) {
    const float* x    = (const float*)d_in[0];
    const float* rw   = (const float*)d_in[1];
    const float* rb   = (const float*)d_in[2];
    const float* wg_s = (const float*)d_in[3];
    const float* wu_s = (const float*)d_in[4];
    const float* wd_s = (const float*)d_in[5];
    const float* wg_l = (const float*)d_in[6];
    const float* wu_l = (const float*)d_in[7];
    const float* wd_l = (const float*)d_in[8];

    char* ws = (char*)d_ws;
    // bf16 workspace layout
    size_t off = 0;
    bf16_t* xb  = (bf16_t*)(ws + off); off += (size_t)TOKENS * DIMK * 2;   // 32 MB
    bf16_t* gsb = (bf16_t*)(ws + off); off += (size_t)HSMALL * DIMK * 2;   //  8 MB
    bf16_t* usb = (bf16_t*)(ws + off); off += (size_t)HSMALL * DIMK * 2;
    bf16_t* dsb = (bf16_t*)(ws + off); off += (size_t)DIMK * HSMALL * 2;
    bf16_t* glb = (bf16_t*)(ws + off); off += (size_t)HLARGE * DIMK * 2;   // 32 MB
    bf16_t* ulb = (bf16_t*)(ws + off); off += (size_t)HLARGE * DIMK * 2;
    bf16_t* dlb = (bf16_t*)(ws + off); off += (size_t)DIMK * HLARGE * 2;
    int* is_large  = (int*)(ws + off); off += (size_t)TOKENS * 4;
    int* small_idx = (int*)(ws + off); off += (size_t)TOKENS * 4;
    int* large_idx = (int*)(ws + off); off += (size_t)CAP * 4;
    int* counts    = (int*)(ws + off); off += 64;

    float* out      = (float*)d_out;
    int*   out_tail = (int*)d_out + (size_t)TOKENS * DIMK;

    // 1) convert everything to bf16
    auto cvt = [&](const float* s, bf16_t* d, int n) {
        int blocks = (n / 4 + 255) / 256;
        cvt_kernel<<<blocks, 256, 0, stream>>>(s, (unsigned short*)d, n);
    };
    cvt(x,    xb,  TOKENS * DIMK);
    cvt(wg_s, gsb, HSMALL * DIMK);
    cvt(wu_s, usb, HSMALL * DIMK);
    cvt(wd_s, dsb, DIMK * HSMALL);
    cvt(wg_l, glb, HLARGE * DIMK);
    cvt(wu_l, ulb, HLARGE * DIMK);
    cvt(wd_l, dlb, DIMK * HLARGE);

    // 2) route
    router_kernel<<<TOKENS / 8, 256, 0, stream>>>(x, rw, rb, is_large);

    // 3) capacity scan + compaction (+ scalar outputs)
    scan_kernel<<<1, 256, 0, stream>>>(is_large, small_idx, large_idx,
                                       counts, out_tail);

    // 4) experts over compacted lists (worst-case grids, device-count exit)
    ffn_kernel<HSMALL><<<TOKENS / 32, 1024, 0, stream>>>(
        xb, gsb, usb, dsb, small_idx, &counts[0], out);
    ffn_kernel<HLARGE><<<CAP / 32, 1024, 0, stream>>>(
        xb, glb, ulb, dlb, large_idx, &counts[1], out);
}
